// HeroEvolver_26946624815513
// MI455X (gfx1250) — matrix operand order repaired and checked by reading, not compiled, not run on hardware
//
#include <hip/hip_runtime.h>

// MI455X (gfx1250, wave32) batched tiny-network forward.
// Every shared-weight linear layer is a WMMA GEMM with batch rows as the
// N dimension: D = W (A-matrix, 16 out-features x K) x Act^T (B-matrix).
// D is produced transposed (lane = batch row, VGPRs = 8 contiguous features),
// so activation writeback is a single ds_store_b128 per lane and biases are
// baked into the GEMM via a 1.0 activation column. Softmax cores on VALU.

typedef __attribute__((ext_vector_type(16))) _Float16 v16h;
typedef __attribute__((ext_vector_type(8)))  _Float16 v8h;
typedef __attribute__((ext_vector_type(8)))  float    v8f;

#define NLAYER 24
// d_in index of each weight matrix (bias is +1)
__constant__ int c_WIDX[NLAYER] = {  3,  5,  7,  9,   // wsa  q,k,v,o
                                    11, 13, 15, 17,   // w2p  q,k,v,o
                                    19, 21, 23, 25,   // p2w  q,k,v,o
                                    27, 29, 31, 33,   // psa  q,k,v,o
                                    35, 37,           // weapon_mlp 1,2
                                    39, 41, 43,       // hero_mlp 1,2,3
                                    45, 47, 49 };     // adv_mlp 1,2,3
__constant__ int c_KARR[NLAYER] = { 15,15,15,15, 15,16,16,15, 12,12,12,12,
                                    12,12,12,12, 55,15, 26,15,12, 22,10,10 };
__constant__ int c_OARR[NLAYER] = { 15,15,15,15, 15,15,15,15, 12,12,12,12,
                                    12,12,12,12, 15,12, 15,12,16, 10,10,1 };
// bias column (activation holds 1.0 there); L16/L21 use special remaps
__constant__ int c_BC[NLAYER]   = { 15,15,15,15, 15,16,16,15, 16,15,15,15,
                                    16,16,16,15, 15,15, 26,15,15, 15,15,15 };

// LDS weight offsets (f16 elems): 16 x Kpad rows; Kpad=32 except L16 (64).
#define WOFF(L) (((L) <= 16) ? (L)*512 : 9216 + ((L)-17)*512)
#define WELEMS  12800

struct HeroArgs { const void* p[51]; };

// ---------------- WMMA fragment helpers (wave32 layouts per CDNA5 ISA) -----

__device__ inline v16h frag_cat(v8h lo, v8h hi) {
  return __builtin_shufflevector(lo, hi, 0, 1, 2, 3, 4, 5, 6, 7,
                                 8, 9, 10, 11, 12, 13, 14, 15);
}

// Weight A-fragment: rows = out-features, stored row-major [16][kpad] in LDS.
// Lane (f=lane&15, h): runs [k0+8h, +8) and [k0+16+8h, +8) of row f.
__device__ inline v16h load_awt(const _Float16* w, int kpad, int k0) {
  const int lane = threadIdx.x & 31;
  const int f = lane & 15, h = lane >> 4;
  const _Float16* p = w + f * kpad + 8 * h + k0;
  return frag_cat(*(const v8h*)p, *(const v8h*)(p + 16));
}

// Activation B-fragment (Act^T): lane (m=lane&15, h): runs [k0+8h, +8) and [k0+16+8h, +8) of row m.
// For 16-wide buffers the second run (k0+16+8h) reads the next row; those K positions hit
// all-zero weight columns, and all overrun targets are finite.
__device__ inline v16h load_bact(const _Float16* act, int lda, int k0) {
  const int lane = threadIdx.x & 31;
  const int m = lane & 15, h = lane >> 4;
  const _Float16* p = act + m * lda + 8 * h + k0;
  return frag_cat(*(const v8h*)p, *(const v8h*)(p + 16));
}

template <int CH>  // K chunks (1 or 2)
__device__ inline v8f gemm_t(const _Float16* w, int kpad,
                             const _Float16* act, int lda) {
  v8f c = {0.f, 0.f, 0.f, 0.f, 0.f, 0.f, 0.f, 0.f};
#pragma unroll
  for (int ch = 0; ch < CH; ++ch) {
    v16h a = load_awt(w, kpad, ch * 32);
    v16h b = load_bact(act, lda, ch * 32);
    c = __builtin_amdgcn_wmma_f32_16x16x32_f16(false, a, false, b, (short)0, c,
                                               false, false);
  }
  return c;
}

// D writeback: lane (m=lane&15, h) holds features f=8h..8h+7 of batch row m.
// Writes Act[m][col0+8h ..] as one b128. 'one' forces col15 := 1.0 (bias col).
__device__ inline void store_d_lds(v8f c, _Float16* act, int ldo, int col0,
                                   bool one, bool relu) {
  const int lane = threadIdx.x & 31;
  const int m = lane & 15, h = lane >> 4;
  v8h o;
#pragma unroll
  for (int r = 0; r < 8; ++r) {
    float v = c[r];
    if (relu) v = fmaxf(v, 0.0f);
    o[r] = (_Float16)v;
  }
  if (one && h == 1) o[7] = (_Float16)1.0f;
  *(v8h*)(act + m * ldo + col0 + 8 * h) = o;
}

// D writeback to global f32: lane writes its 8 contiguous features (masked N).
__device__ inline void store_d_global(v8f c, float* obase, int col0, int N) {
  const int lane = threadIdx.x & 31;
  const int m = lane & 15, h = lane >> 4;
  float* p = obase + (long)m * 130 + col0 + 8 * h;
#pragma unroll
  for (int r = 0; r < 8; ++r)
    if (8 * h + r < N) p[r] = c[r];
}

__device__ inline float getf(v8h lo, v8h hi, int i) {
  return (i < 8) ? (float)lo[i & 7] : (float)hi[i & 7];
}

// Tiny attention core: lanes 0..15 each own one batch row; buffers are
// [token][16][16] f16. Full-row b128 traffic; writes softmax(QK^T/s)V back
// into Qb rows; unused tail cols zeroed, col15 := 1.0 (bias col for o-proj).
template <int LQ, int LK, int NH, int HD>
__device__ inline void attn_inplace(_Float16* Qb, const _Float16* Kb,
                                    const _Float16* Vb, float scale) {
  const int lane = threadIdx.x & 31;
  if (lane < 16) {
    const int m = lane;
    v8h kl[LK], kh[LK], vl[LK], vh[LK];
#pragma unroll
    for (int u = 0; u < LK; ++u) {
      const _Float16* kp = Kb + (u * 16 + m) * 16;
      const _Float16* vp = Vb + (u * 16 + m) * 16;
      kl[u] = *(const v8h*)kp; kh[u] = *(const v8h*)(kp + 8);
      vl[u] = *(const v8h*)vp; vh[u] = *(const v8h*)(vp + 8);
    }
#pragma unroll
    for (int t = 0; t < LQ; ++t) {
      _Float16* qp = Qb + (t * 16 + m) * 16;
      v8h q0 = *(const v8h*)qp, q1 = *(const v8h*)(qp + 8);
      float qf[16];
#pragma unroll
      for (int d = 0; d < 8; ++d) { qf[d] = (float)q0[d]; qf[8 + d] = (float)q1[d]; }
      _Float16 orow[16];
#pragma unroll
      for (int d = 0; d < 15; ++d) orow[d] = (_Float16)0.0f;
      orow[15] = (_Float16)1.0f;
#pragma unroll
      for (int h2 = 0; h2 < NH; ++h2) {
        const int d0 = h2 * HD;
        float sc[LK], mx = -3.0e38f;
#pragma unroll
        for (int u = 0; u < LK; ++u) {
          float s = 0.f;
#pragma unroll
          for (int d = 0; d < HD; ++d)
            s += qf[d0 + d] * getf(kl[u], kh[u], d0 + d);
          s *= scale;
          sc[u] = s;
          mx = fmaxf(mx, s);
        }
        float den = 0.f;
#pragma unroll
        for (int u = 0; u < LK; ++u) { sc[u] = __expf(sc[u] - mx); den += sc[u]; }
        const float inv = 1.0f / den;
#pragma unroll
        for (int d = 0; d < HD; ++d) {
          float o = 0.f;
#pragma unroll
          for (int u = 0; u < LK; ++u) o += sc[u] * getf(vl[u], vh[u], d0 + d);
          orow[d0 + d] = (_Float16)(o * inv);
        }
      }
      v8h o0, o1;
#pragma unroll
      for (int d = 0; d < 8; ++d) { o0[d] = orow[d]; o1[d] = orow[8 + d]; }
      ((v8h*)qp)[0] = o0;
      ((v8h*)qp)[1] = o1;
    }
  }
}

// weapon_mlp layer1 weight remap: mi cols [x 0-14 |1.0@15| sa 16-30 |0@31|
// ca 32-46 |0@47| act 48-57 | 0]; original input order [x15|sa15|ca15|act10].
__device__ inline float wv_weapon(const float* w, const float* b, int f, int k) {
  if (k < 15)             return w[f * 55 + k];
  if (k == 15)            return b[f];
  if (k >= 16 && k <= 30) return w[f * 55 + (k - 1)];
  if (k >= 32 && k <= 46) return w[f * 55 + (k - 2)];
  if (k >= 48 && k <= 57) return w[f * 55 + (k - 3)];
  return 0.f;
}
// adv_mlp layer1 remap: cols [comb 0-11 |0,0,0| 1.0@15 | act 16-25 | 0]
__device__ inline float wv_adv1(const float* w, const float* b, int f, int k) {
  if (k < 12)             return w[f * 22 + k];
  if (k == 15)            return b[f];
  if (k >= 16 && k <= 25) return w[f * 22 + (k - 4)];
  return 0.f;
}

// ---------------------------------------------------------------------------

__global__ void __launch_bounds__(64)
hero_forward(HeroArgs A, float* __restrict__ out) {
  __shared__ __align__(16) _Float16 Wl[WELEMS];      // weights [16][Kpad] f16
  __shared__ float    Pos[12];                       // pos_emb (4x3)
  __shared__ __align__(16) _Float16 wbuf[2][7680];   // per-wave tiles
  __shared__ float    ahb[2][16];                    // per-wave ah scalars

  const int tid = threadIdx.x;

  // ---- stage weights (f32 -> f16, row-major [out-feature][Kpad], bias col)
  for (int L = 0; L < NLAYER; ++L) {
    const float* wsrc = (const float*)A.p[c_WIDX[L]];
    const float* bsrc = (const float*)A.p[c_WIDX[L] + 1];
    const int K = c_KARR[L], O = c_OARR[L], BC = c_BC[L];
    const int sh = (L == 16) ? 6 : 5;
    const int nel = 16 << sh;
    const int off = WOFF(L);
    for (int i = tid; i < nel; i += 64) {
      const int f = i >> sh, k = i & ((1 << sh) - 1);
      float v = 0.f;
      if (f < O) {
        if (L == 16)      v = wv_weapon(wsrc, bsrc, f, k);
        else if (L == 21) v = wv_adv1(wsrc, bsrc, f, k);
        else if (k < K)   v = wsrc[f * K + k];
        else if (k == BC) v = bsrc[f];
      }
      Wl[off + i] = (_Float16)v;
    }
  }
  if (tid < 12) Pos[tid] = ((const float*)A.p[2])[tid];
  __syncthreads();

#define WP(L) (Wl + WOFF(L))

  const int wave = tid >> 5;
  const int lane = tid & 31;
  const int m    = lane & 15;
  const int hf   = lane >> 4;
  const long base = ((long)blockIdx.x * 2 + wave) * 16;

  _Float16* wb  = wbuf[wave];
  _Float16* xb  = wb;           // [4][16][16]  x tokens (hwt|pos|1.0@15)
  _Float16* pt0 = wb + 1024;    // [16][32]     ht | 1.0@16 | zeros
  _Float16* pt1 = wb + 1536;    // [16][32]     at | 1.0@16 | zeros
  _Float16* QB  = wb + 2048;    // [4][16][16]
  _Float16* KB  = wb + 3072;    // [4][16][16]
  _Float16* VB  = wb + 4096;    // [4][16][16]
  _Float16* SAb = wb + 5120;    // [4][16][16]  wsa output
  _Float16* TA  = wb + 6144;    // [16][64]     generic GEMM staging (mi etc.)
  _Float16* TB  = wb + 7168;    // [16][32]     MLP hidden staging
  float* ahw = ahb[wave];

  const float* st = (const float*)A.p[0] + (base + m) * 130;
  const float* ac = (const float*)A.p[1] + (base + m) * 10;
  float* orow  = out + (base + m) * 130;
  float* obase = out + base * 130;

  float actr[10];
#pragma unroll
  for (int i = 0; i < 10; ++i) actr[i] = 0.f;

  // ---- phase 0: zero pt/TA/TB regions, stage inputs, pass-through cols ----
  {
    v8h z = {0, 0, 0, 0, 0, 0, 0, 0};
    for (int i = 128 + lane; i < 256; i += 32) ((v8h*)wb)[i] = z;  // pt0,pt1
    for (int i = 768 + lane; i < 960; i += 32) ((v8h*)wb)[i] = z;  // TA,TB
  }
  if (hf == 0) {
    orow[0] = st[0];                                   // hh
#pragma unroll
    for (int e = 0; e < 16; ++e) pt0[m * 32 + e] = (_Float16)st[1 + e];
    pt0[m * 32 + 16] = (_Float16)1.0f;                 // bias col
#pragma unroll
    for (int t = 0; t < 4; ++t) {
#pragma unroll
      for (int e = 0; e < 12; ++e)
        xb[(t * 16 + m) * 16 + e] = (_Float16)st[17 + 12 * t + e];
#pragma unroll
      for (int e = 0; e < 3; ++e)
        xb[(t * 16 + m) * 16 + 12 + e] = (_Float16)Pos[t * 3 + e];
      xb[(t * 16 + m) * 16 + 15] = (_Float16)1.0f;     // bias col
    }
#pragma unroll
    for (int i = 0; i < 10; ++i) actr[i] = ac[i];
  } else {
    ahw[m] = st[65];
#pragma unroll
    for (int e = 0; e < 16; ++e) {                     // at (+ pass-through)
      const float v = st[66 + e];
      pt1[m * 32 + e] = (_Float16)v;
      orow[66 + e] = v;
    }
    pt1[m * 32 + 16] = (_Float16)1.0f;                 // bias col
#pragma unroll
    for (int e = 0; e < 48; ++e) orow[82 + e] = st[82 + e];  // awt
  }
  __syncthreads();

  // ---- wsa: self-attn over 4 weapon tokens (E=15, 3 heads, hd=5) ----
  for (int t = 0; t < 4; ++t) {
    store_d_lds(gemm_t<1>(WP(0), 32, xb + t * 256, 16), QB + t * 256, 16, 0, false, false);
    store_d_lds(gemm_t<1>(WP(1), 32, xb + t * 256, 16), KB + t * 256, 16, 0, false, false);
    store_d_lds(gemm_t<1>(WP(2), 32, xb + t * 256, 16), VB + t * 256, 16, 0, false, false);
  }
  __syncthreads();
  attn_inplace<4, 4, 3, 5>(QB, KB, VB, 0.44721360f);   // 1/sqrt(5)
  __syncthreads();
  for (int t = 0; t < 4; ++t)
    store_d_lds(gemm_t<1>(WP(3), 32, QB + t * 256, 16), SAb + t * 256, 16, 0, false, false);
  __syncthreads();

  // ---- w2p: x attends over pt=[ht,at] (E=15, kd=16, 3 heads, hd=5) ----
  for (int t = 0; t < 4; ++t)
    store_d_lds(gemm_t<1>(WP(4), 32, xb + t * 256, 16), QB + t * 256, 16, 0, false, false);
  store_d_lds(gemm_t<1>(WP(5), 32, pt0, 32), KB,       16, 0, false, false);
  store_d_lds(gemm_t<1>(WP(5), 32, pt1, 32), KB + 256, 16, 0, false, false);
  store_d_lds(gemm_t<1>(WP(6), 32, pt0, 32), VB,       16, 0, false, false);
  store_d_lds(gemm_t<1>(WP(6), 32, pt1, 32), VB + 256, 16, 0, false, false);
  __syncthreads();
  attn_inplace<4, 2, 3, 5>(QB, KB, VB, 0.44721360f);
  __syncthreads();

  // ---- weapon MLP per token: mi (cols 0..57) -> 15 relu -> 12 ----
  if (hf == 0) {
#pragma unroll
    for (int i = 0; i < 10; ++i) TA[m * 64 + 48 + i] = (_Float16)actr[i];
  }
  for (int t = 0; t < 4; ++t) {
    if (hf == 0) {
      const v8h* xr = (const v8h*)(xb + (t * 16 + m) * 16);    // x|1.0@15
      ((v8h*)(TA + m * 64))[0] = xr[0];
      ((v8h*)(TA + m * 64))[1] = xr[1];
      const v8h* sr = (const v8h*)(SAb + (t * 16 + m) * 16);   // sa|0@31
      ((v8h*)(TA + m * 64))[2] = sr[0];
      ((v8h*)(TA + m * 64))[3] = sr[1];
    }
    store_d_lds(gemm_t<1>(WP(7), 32, QB + t * 256, 16), TA, 64, 32, false, false); // ca
    __syncthreads();
    store_d_lds(gemm_t<2>(WP(16), 64, TA, 64), TB, 32, 0, true, true);
    __syncthreads();
    store_d_global(gemm_t<1>(WP(17), 32, TB, 32), obase, 17 + 12 * t, 12);
    __syncthreads();
  }

  // ---- hero MLP: [ht|act|1.0@26] -> 15 relu -> 12 relu -> 16 ----
  if (hf == 0) {
    const v8h* hr = (const v8h*)(pt0 + m * 32);
    ((v8h*)(TA + m * 64))[0] = hr[0];
    ((v8h*)(TA + m * 64))[1] = hr[1];
#pragma unroll
    for (int i = 0; i < 10; ++i) TA[m * 64 + 16 + i] = (_Float16)actr[i];
    TA[m * 64 + 26] = (_Float16)1.0f;                  // bias col (BC=26)
  }
  __syncthreads();
  store_d_lds(gemm_t<1>(WP(18), 32, TA, 64), TB, 32, 0, true, true);
  __syncthreads();
  store_d_lds(gemm_t<1>(WP(19), 32, TB, 32), TB, 32, 0, true, true);
  __syncthreads();
  store_d_global(gemm_t<1>(WP(20), 32, TB, 32), obase, 1, 16);
  __syncthreads();

  // ---- p2w: at[:12] attends over hwt (E=12, 2 heads, hd=6) ----
  store_d_lds(gemm_t<1>(WP(8), 32, pt1, 32), QB, 16, 0, false, false);
  for (int t = 0; t < 4; ++t) {
    store_d_lds(gemm_t<1>(WP(9),  32, xb + t * 256, 16), KB + t * 256, 16, 0, false, false);
    store_d_lds(gemm_t<1>(WP(10), 32, xb + t * 256, 16), VB + t * 256, 16, 0, false, false);
  }
  __syncthreads();
  attn_inplace<1, 4, 2, 6>(QB, KB, VB, 0.40824829f);   // 1/sqrt(6)
  __syncthreads();
  v8f cca2 = gemm_t<1>(WP(11), 32, QB, 16);            // ca2 (bias baked)
  __syncthreads();

  // ---- psa: self-attn over px=[ht[:12],at[:12]] (E=12, 2 heads, hd=6) ----
  store_d_lds(gemm_t<1>(WP(12), 32, pt0, 32), QB,       16, 0, false, false);
  store_d_lds(gemm_t<1>(WP(12), 32, pt1, 32), QB + 256, 16, 0, false, false);
  store_d_lds(gemm_t<1>(WP(13), 32, pt0, 32), KB,       16, 0, false, false);
  store_d_lds(gemm_t<1>(WP(13), 32, pt1, 32), KB + 256, 16, 0, false, false);
  store_d_lds(gemm_t<1>(WP(14), 32, pt0, 32), VB,       16, 0, false, false);
  store_d_lds(gemm_t<1>(WP(14), 32, pt1, 32), VB + 256, 16, 0, false, false);
  __syncthreads();
  attn_inplace<2, 2, 2, 6>(QB, KB, VB, 0.40824829f);
  __syncthreads();
  v8f cs0 = gemm_t<1>(WP(15), 32, QB,       16);
  v8f cs1 = gemm_t<1>(WP(15), 32, QB + 256, 16);

  // fragment-level fusion: 0.5*(ca2 + mean(sa2 tokens)); biases already baked
  {
    v8h o;
#pragma unroll
    for (int r = 0; r < 8; ++r)
      o[r] = (_Float16)(0.5f * (cca2[r] + 0.5f * (cs0[r] + cs1[r])));
    if (hf == 1) o[7] = (_Float16)1.0f;                // col15 = bias col
    *(v8h*)(TA + m * 64 + 8 * hf) = o;
  }
  if (hf == 0) {
#pragma unroll
    for (int i = 0; i < 10; ++i) TA[m * 64 + 16 + i] = (_Float16)actr[i];
  }
  __syncthreads();

  // ---- adv MLP: (22) -> 10 relu -> 10 relu -> 1 ; ah_pred = ah + out ----
  store_d_lds(gemm_t<1>(WP(21), 32, TA, 64), TB, 32, 0, true, true);
  __syncthreads();
  store_d_lds(gemm_t<1>(WP(22), 32, TB, 32), TB, 32, 0, true, true);
  __syncthreads();
  v8f cah = gemm_t<1>(WP(23), 32, TB, 32);
  if (hf == 0)
    orow[65] = ahw[m] + cah[0];                        // D row0 = adv output
}

extern "C" void kernel_launch(void* const* d_in, const int* in_sizes, int n_in,
                              void* d_out, int out_size, void* d_ws, size_t ws_size,
                              hipStream_t stream) {
  (void)out_size; (void)d_ws; (void)ws_size;
  HeroArgs a;
  for (int i = 0; i < 51; ++i) a.p[i] = (i < n_in) ? d_in[i] : nullptr;
  const int B = in_sizes[0] / 130;        // 131072
  const int grid = B / 32;                // 2 waves x 16 rows per block
  hipLaunchKernelGGL(hero_forward, dim3(grid), dim3(64), 0, stream,
                     a, (float*)d_out);
}
